// GPT_25237227832043
// MI455X (gfx1250) — matrix operand order repaired and checked by reading, not compiled, not run on hardware
//
#include <hip/hip_runtime.h>
#include <hip/hip_bf16.h>
#include <math.h>

// ---------------------------------------------------------------------------
// GPT forward for gfx1250 (MI455X): bf16 WMMA GEMMs (v_wmma_f32_16x16x32_bf16),
// double-buffered LDS staging with async global->LDS copies (ASYNCcnt),
// flash-style GQA attention, fused epilogues.
// ---------------------------------------------------------------------------

#ifndef USE_ASYNC_LDS
#define USE_ASYNC_LDS 1
#endif

typedef __bf16 bf16_t;
typedef __attribute__((ext_vector_type(16))) __bf16 v16bf;
typedef __attribute__((ext_vector_type(8)))  __bf16 v8bf;
typedef __attribute__((ext_vector_type(4)))  __bf16 v4bf;
typedef __attribute__((ext_vector_type(8)))  float  v8f;

#define V_SZ   50257
#define D_SZ   1024
#define H_SZ   16
#define G_SZ   4
#define NL_SZ  8
#define S_SZ   1024
#define HD_SZ  64
#define KVD_SZ 256
#define FF_SZ  4096
#define M_SZ   2048          // B * S
#define LEPS   1e-5f
#define WTET_LD 50264        // V padded to multiple of 8 (16B-aligned bf16 rows)

__device__ __forceinline__ v16bf make_frag(v8bf lo, v8bf hi) {
  return __builtin_shufflevector(lo, hi, 0,1,2,3,4,5,6,7,8,9,10,11,12,13,14,15);
}
__device__ __forceinline__ v8f wmma_bf16(v16bf a, v16bf b, v8f c) {
  return __builtin_amdgcn_wmma_f32_16x16x32_bf16(false, a, false, b, (short)0, c,
                                                 false, false);
}

#if USE_ASYNC_LDS
// One 16B lane-granular async copy global -> LDS (tracked by ASYNCcnt).
__device__ __forceinline__ void async_ld_b128(unsigned lds_off, const void* g) {
  asm volatile("global_load_async_to_lds_b128 %0, %1, off"
               :: "v"(lds_off), "v"((unsigned long long)(size_t)g) : "memory");
}
__device__ __forceinline__ void wait_async0() {
  asm volatile("s_wait_asynccnt 0x0" ::: "memory");
}
#endif

// ------------------------------- embedding ---------------------------------
__global__ void k_embed(const int* __restrict__ idx, const float* __restrict__ wte,
                        const float* __restrict__ wpe, float* __restrict__ x) {
  int row = blockIdx.x;                 // 0..2047 = b*S + s
  int s   = row & (S_SZ - 1);
  int tok = idx[row];
  const float* wt = wte + (size_t)tok * D_SZ;
  const float* wp = wpe + (size_t)s   * D_SZ;
  float* xo = x + (size_t)row * D_SZ;
  for (int i = threadIdx.x; i < D_SZ; i += 256) xo[i] = wt[i] + wp[i];
}

// ------------------------------- layernorm ---------------------------------
__global__ void k_layernorm(const float* __restrict__ x, const float* __restrict__ g,
                            const float* __restrict__ b, bf16_t* __restrict__ out) {
  __shared__ float red[256];
  int row = blockIdx.x;
  const float* xr = x + (size_t)row * D_SZ;
  float v[4]; float lsum = 0.f;
  #pragma unroll
  for (int i = 0; i < 4; i++) { v[i] = xr[threadIdx.x + i * 256]; lsum += v[i]; }
  red[threadIdx.x] = lsum; __syncthreads();
  for (int st = 128; st > 0; st >>= 1) {
    if (threadIdx.x < st) red[threadIdx.x] += red[threadIdx.x + st];
    __syncthreads();
  }
  float mu = red[0] * (1.0f / D_SZ);
  __syncthreads();
  float lvar = 0.f;
  #pragma unroll
  for (int i = 0; i < 4; i++) { float d = v[i] - mu; lvar += d * d; }
  red[threadIdx.x] = lvar; __syncthreads();
  for (int st = 128; st > 0; st >>= 1) {
    if (threadIdx.x < st) red[threadIdx.x] += red[threadIdx.x + st];
    __syncthreads();
  }
  float rs = rsqrtf(red[0] * (1.0f / D_SZ) + LEPS);
  #pragma unroll
  for (int i = 0; i < 4; i++) {
    int c = threadIdx.x + i * 256;
    out[(size_t)row * D_SZ + c] = (bf16_t)((v[i] - mu) * rs * g[c] + b[c]);
  }
}

// ----------------------- wte^T (bf16) for the LM head ----------------------
__global__ void k_wteT(const float* __restrict__ wte, bf16_t* __restrict__ wteT) {
  __shared__ float tile[32][33];
  int n0 = blockIdx.x * 32;   // vocab
  int k0 = blockIdx.y * 32;   // dim
  int tx = threadIdx.x, ty = threadIdx.y;      // block (32,8)
  #pragma unroll
  for (int i = 0; i < 4; i++) {
    int n = n0 + ty + i * 8;
    tile[ty + i * 8][tx] = (n < V_SZ) ? wte[(size_t)n * D_SZ + k0 + tx] : 0.0f;
  }
  __syncthreads();
  #pragma unroll
  for (int i = 0; i < 4; i++) {
    int k = k0 + ty + i * 8;
    int n = n0 + tx;
    if (n < WTET_LD) wteT[(size_t)k * WTET_LD + n] = (bf16_t)tile[tx][ty + i * 8];
  }
}

// --------------------- K repack: kT[b][g][s][d] ----------------------------
__global__ void k_transpose_k(const bf16_t* __restrict__ kb, bf16_t* __restrict__ kT) {
  int fl = blockIdx.x * 256 + threadIdx.x;     // 2048*256 threads
  int sg = fl >> 8;                            // b*S + s
  int c  = fl & 255;                           // g*64 + d
  int b  = sg >> 10, s = sg & 1023;
  kT[((size_t)(b * G_SZ + (c >> 6)) * S_SZ + s) * HD_SZ + (c & 63)] =
      kb[(size_t)sg * KVD_SZ + c];
}

// ------------------------------- WMMA GEMM ---------------------------------
// C[M,N] = act(A_bf16[M,K] @ B[K,N] + bias). Block tile 128x128, BK=32,
// ping-pong LDS buffers, async A staging, one barrier per K-step.
enum { EPI_BF16 = 0, EPI_GELU = 1, EPI_RES = 2, EPI_F32 = 3 };

template <int EPI, bool BBF16>
__global__ __launch_bounds__(256) void k_gemm(
    const bf16_t* __restrict__ A, int lda,
    const void* __restrict__ Bp, int ldb,
    const float* __restrict__ bias,
    const float* __restrict__ res, int ldr,
    void* __restrict__ Outp, int ldo,
    int N, int K) {
  __shared__ bf16_t As[2][128 * 40];   // +8 halves pad -> conflict-free frags
  __shared__ bf16_t Bs[2][32 * 136];
  const int tid  = threadIdx.x;
  const int lane = tid & 31, wave = tid >> 5;
  const int m0 = blockIdx.y * 128;
  const int n0 = blockIdx.x * 128;
  const int waveM = (wave >> 2) * 64;
  const int waveN = (wave & 3) * 32;
  const int lrow = lane & 15;
  const int lhi  = lane >> 4;

  v8f acc[4][2] = {};
  float4 pbf[4];
  v8bf   pbb[2];
#if !USE_ASYNC_LDS
  v8bf   pa[2];
#endif

  auto loadA = [&](int k0, int buf) {
    #pragma unroll
    for (int it = 0; it < 2; it++) {
      int seg = tid + it * 256;
      int r = seg >> 2, c8 = (seg & 3) * 8;
      const bf16_t* g = A + (size_t)(m0 + r) * lda + k0 + c8;
#if USE_ASYNC_LDS
      async_ld_b128((unsigned)(size_t)(&As[buf][r * 40 + c8]), g);
#else
      (void)buf;
      pa[it] = *(const v8bf*)g;
#endif
    }
  };
  auto storeA = [&](int buf) {
#if !USE_ASYNC_LDS
    #pragma unroll
    for (int it = 0; it < 2; it++) {
      int seg = tid + it * 256;
      int r = seg >> 2, c8 = (seg & 3) * 8;
      *(v8bf*)(&As[buf][r * 40 + c8]) = pa[it];
    }
#else
    (void)buf;
#endif
  };
  auto loadB = [&](int k0) {
    if constexpr (BBF16) {
      const bf16_t* B = (const bf16_t*)Bp;
      #pragma unroll
      for (int it = 0; it < 2; it++) {
        int seg = tid + it * 256;
        int r = seg >> 4, c8 = (seg & 15) * 8;
        int gc = n0 + c8;
        if (gc + 8 <= ldb) {
          pbb[it] = *(const v8bf*)(B + (size_t)(k0 + r) * ldb + gc);
        } else {
          #pragma unroll
          for (int e = 0; e < 8; e++)
            pbb[it][e] = (gc + e < ldb) ? B[(size_t)(k0 + r) * ldb + gc + e]
                                        : (bf16_t)0.0f;
        }
      }
    } else {
      const float* B = (const float*)Bp;
      #pragma unroll
      for (int it = 0; it < 4; it++) {
        int seg = tid + it * 256;
        int r = seg >> 5, c4 = (seg & 31) * 4;
        pbf[it] = *(const float4*)(B + (size_t)(k0 + r) * ldb + n0 + c4);
      }
    }
  };
  auto storeB = [&](int buf) {
    if constexpr (BBF16) {
      #pragma unroll
      for (int it = 0; it < 2; it++) {
        int seg = tid + it * 256;
        int r = seg >> 4, c8 = (seg & 15) * 8;
        *(v8bf*)(&Bs[buf][r * 136 + c8]) = pbb[it];
      }
    } else {
      #pragma unroll
      for (int it = 0; it < 4; it++) {
        int seg = tid + it * 256;
        int r = seg >> 5, c4 = (seg & 31) * 4;
        v4bf t;
        t[0] = (bf16_t)pbf[it].x; t[1] = (bf16_t)pbf[it].y;
        t[2] = (bf16_t)pbf[it].z; t[3] = (bf16_t)pbf[it].w;
        *(v4bf*)(&Bs[buf][r * 136 + c4]) = t;       // one b64 DS store
      }
    }
  };

  // prologue: stage tile 0
  loadA(0, 0); loadB(0);
  storeA(0);   storeB(0);
#if USE_ASYNC_LDS
  wait_async0();
#endif
  __syncthreads();

  int cur = 0;
  for (int k0 = 0; k0 < K; k0 += 32) {
    const bool hn = (k0 + 32) < K;
    if (hn) { loadA(k0 + 32, cur ^ 1); loadB(k0 + 32); }   // prefetch next tile

    // ---- fragments + 8 WMMAs from buffer `cur` ----
    v16bf af[4];
    #pragma unroll
    for (int i = 0; i < 4; i++) {
      const bf16_t* p = &As[cur][(waveM + i * 16 + lrow) * 40 + lhi * 8];
      af[i] = make_frag(*(const v8bf*)p, *(const v8bf*)(p + 16));
    }
    v16bf bfr[2];
    {
      #pragma unroll
      for (int j = 0; j < 2; j++) {
        #pragma unroll
        for (int e = 0; e < 16; e++) {
          int kk = (e & 7) + lhi * 8 + ((e >> 3) << 4);
          bfr[j][e] = Bs[cur][kk * 136 + waveN + j * 16 + lrow];
        }
      }
    }
    #pragma unroll
    for (int i = 0; i < 4; i++)
      #pragma unroll
      for (int j = 0; j < 2; j++)
        acc[i][j] = wmma_bf16(af[i], bfr[j], acc[i][j]);

    if (hn) { storeA(cur ^ 1); storeB(cur ^ 1); }
#if USE_ASYNC_LDS
    wait_async0();
#endif
    __syncthreads();
    cur ^= 1;
  }

  // ---- epilogue ----
  #pragma unroll
  for (int i = 0; i < 4; i++) {
    #pragma unroll
    for (int j = 0; j < 2; j++) {
      int gcol = n0 + waveN + j * 16 + lrow;
      float bv = bias ? bias[gcol] : 0.0f;
      #pragma unroll
      for (int r = 0; r < 8; r++) {
        int grow = m0 + waveM + i * 16 + r + lhi * 8;
        float val = acc[i][j][r] + bv;
        if constexpr (EPI == EPI_BF16) {
          ((bf16_t*)Outp)[(size_t)grow * ldo + gcol] = (bf16_t)val;
        } else if constexpr (EPI == EPI_GELU) {
          float gl = 0.5f * val * (1.0f + erff(val * 0.70710678118654752f));
          ((bf16_t*)Outp)[(size_t)grow * ldo + gcol] = (bf16_t)gl;
        } else if constexpr (EPI == EPI_RES) {
          float* o = (float*)Outp;
          o[(size_t)grow * ldo + gcol] = res[(size_t)grow * ldr + gcol] + val;
        } else {
          if (gcol < N) ((float*)Outp)[(size_t)grow * ldo + gcol] = val;
        }
      }
    }
  }
}

// --------------------------- flash attention -------------------------------
// Unmasked softmax over all S keys (reference never applies the causal mask).
// Per wave: 16 q rows, hd=64; stream keys in chunks of 32.
__global__ __launch_bounds__(256) void k_attn(
    const bf16_t* __restrict__ qb,   // [M, D]   (b*S+s, h*64+d)
    const bf16_t* __restrict__ kT,   // [B,G,S,64]
    const bf16_t* __restrict__ vb,   // [M, KVD] (b*S+s, g*64+d)
    bf16_t* __restrict__ att) {      // [M, D]
  __shared__ bf16_t pbuf[8 * 16 * 40];
  const int lane = threadIdx.x & 31, wave = threadIdx.x >> 5;
  const int lrow = lane & 15, lhi = lane >> 4;
  const int qblk = blockIdx.x & 7;          // S/128
  const int bh   = blockIdx.x >> 3;
  const int h = bh & 15, b = bh >> 4;
  const int g = h >> 2;                     // h / (H/G)
  const int q0 = qblk * 128 + wave * 16;

  v16bf qf[2];
  {
    const bf16_t* qr = qb + (size_t)(b * S_SZ + q0 + lrow) * D_SZ + h * HD_SZ + lhi * 8;
    qf[0] = make_frag(*(const v8bf*)qr,        *(const v8bf*)(qr + 16));
    qf[1] = make_frag(*(const v8bf*)(qr + 32), *(const v8bf*)(qr + 48));
  }

  float m[8], l[8];
  v8f o[4] = {};
  #pragma unroll
  for (int r = 0; r < 8; r++) { m[r] = -3.0e38f; l[r] = 0.0f; }

  bf16_t* pw = pbuf + wave * 16 * 40;
  const bf16_t* kTb = kT + (size_t)(b * G_SZ + g) * HD_SZ * S_SZ;
  const bf16_t* vbb = vb + (size_t)b * S_SZ * KVD_SZ + g * HD_SZ;

  for (int kc = 0; kc < S_SZ; kc += 32) {
    v8f s[2] = {};
    #pragma unroll
    for (int j = 0; j < 2; j++) {
      int key0 = kc + j * 16;
      const bf16_t* p0 = kTb + (size_t)(key0 + lrow) * HD_SZ + lhi * 8;
      const bf16_t* p1 = p0 + 32;
      v16bf kb0 = make_frag(*(const v8bf*)p0, *(const v8bf*)(p0 + 16));
      v16bf kb1 = make_frag(*(const v8bf*)p1, *(const v8bf*)(p1 + 16));
      s[j] = wmma_bf16(qf[0], kb0, s[j]);
      s[j] = wmma_bf16(qf[1], kb1, s[j]);
    }
    // scale + online softmax stats (rows in vgpr index; cols across 16 lanes)
    float cm[8];
    #pragma unroll
    for (int r = 0; r < 8; r++) {
      s[0][r] *= 0.125f; s[1][r] *= 0.125f;
      cm[r] = fmaxf(s[0][r], s[1][r]);
    }
    #pragma unroll
    for (int msk = 1; msk < 16; msk <<= 1)
      #pragma unroll
      for (int r = 0; r < 8; r++)
        cm[r] = fmaxf(cm[r], __shfl_xor(cm[r], msk, 32));
    float p0[8], p1[8], rs[8];
    #pragma unroll
    for (int r = 0; r < 8; r++) {
      float mn = fmaxf(m[r], cm[r]);
      float f  = __expf(m[r] - mn);
      m[r] = mn;
      p0[r] = __expf(s[0][r] - mn);
      p1[r] = __expf(s[1][r] - mn);
      rs[r] = p0[r] + p1[r];
      l[r] *= f;
      #pragma unroll
      for (int t = 0; t < 4; t++) o[t][r] *= f;
    }
    #pragma unroll
    for (int msk = 1; msk < 16; msk <<= 1)
      #pragma unroll
      for (int r = 0; r < 8; r++)
        rs[r] += __shfl_xor(rs[r], msk, 32);
    #pragma unroll
    for (int r = 0; r < 8; r++) l[r] += rs[r];
    // C-layout -> A-layout via per-wave LDS bounce (in-order DS within wave)
    #pragma unroll
    for (int r = 0; r < 8; r++) {
      int prow = r + lhi * 8;
      pw[prow * 40 + lrow]      = (bf16_t)p0[r];
      pw[prow * 40 + 16 + lrow] = (bf16_t)p1[r];
    }
    v16bf pf;
    {
      const bf16_t* p = pw + lrow * 40 + lhi * 8;
      pf = make_frag(*(const v8bf*)p, *(const v8bf*)(p + 16));
    }
    // o += p @ v  (V rows are contiguous -> direct B fragments)
    {
      #pragma unroll
      for (int t = 0; t < 4; t++) {
        v16bf vf;
        #pragma unroll
        for (int e = 0; e < 16; e++) {
          int kk = (e & 7) + lhi * 8 + ((e >> 3) << 4);
          vf[e] = vbb[(size_t)(kc + kk) * KVD_SZ + t * 16 + lrow];
        }
        o[t] = wmma_bf16(pf, vf, o[t]);
      }
    }
  }
  #pragma unroll
  for (int r = 0; r < 8; r++) {
    float inv = 1.0f / l[r];
    int grow = b * S_SZ + q0 + r + lhi * 8;
    #pragma unroll
    for (int t = 0; t < 4; t++)
      att[(size_t)grow * D_SZ + h * HD_SZ + t * 16 + lrow] = (bf16_t)(o[t][r] * inv);
  }
}

// ------------------------------- NLL loss ----------------------------------
__global__ void k_loss(const float* __restrict__ logits, const int* __restrict__ tgt,
                       float* __restrict__ loss) {
  __shared__ float smax[256], ssum[256];
  int row = blockIdx.x;
  const float* lr = logits + (size_t)row * V_SZ;
  float mx = -3.0e38f, sm = 0.0f;
  for (int c = threadIdx.x; c < V_SZ; c += 256) {
    float v = lr[c];
    if (v > mx) { sm = sm * __expf(mx - v) + 1.0f; mx = v; }
    else         sm += __expf(v - mx);
  }
  smax[threadIdx.x] = mx; ssum[threadIdx.x] = sm;
  __syncthreads();
  for (int st = 128; st > 0; st >>= 1) {
    if (threadIdx.x < st) {
      float m1 = smax[threadIdx.x], s1 = ssum[threadIdx.x];
      float m2 = smax[threadIdx.x + st], s2 = ssum[threadIdx.x + st];
      float mm = fmaxf(m1, m2);
      smax[threadIdx.x] = mm;
      ssum[threadIdx.x] = s1 * __expf(m1 - mm) + s2 * __expf(m2 - mm);
    }
    __syncthreads();
  }
  if (threadIdx.x == 0) {
    float lse = smax[0] + __logf(ssum[0]);
    atomicAdd(loss, (lse - lr[tgt[row]]) * (1.0f / M_SZ));
  }
}

// ------------------------------- launcher ----------------------------------
extern "C" void kernel_launch(void* const* d_in, const int* in_sizes, int n_in,
                              void* d_out, int out_size, void* d_ws, size_t ws_size,
                              hipStream_t stream) {
  (void)in_sizes; (void)n_in; (void)out_size; (void)ws_size;
  const int*   idx  = (const int*)  d_in[0];
  const int*   tgt  = (const int*)  d_in[1];
  const float* wte  = (const float*)d_in[2];
  const float* wpe  = (const float*)d_in[3];
  const float* ln1g = (const float*)d_in[4];
  const float* ln1b = (const float*)d_in[5];
  const float* wq   = (const float*)d_in[6];
  const float* bq   = (const float*)d_in[7];
  const float* wk   = (const float*)d_in[8];
  const float* bk   = (const float*)d_in[9];
  const float* wv   = (const float*)d_in[10];
  const float* bv   = (const float*)d_in[11];
  const float* wo   = (const float*)d_in[12];
  const float* bo   = (const float*)d_in[13];
  const float* ln2g = (const float*)d_in[14];
  const float* ln2b = (const float*)d_in[15];
  const float* w1   = (const float*)d_in[16];
  const float* b1   = (const float*)d_in[17];
  const float* w2   = (const float*)d_in[18];
  const float* b2   = (const float*)d_in[19];
  const float* lnfg = (const float*)d_in[20];
  const float* lnfb = (const float*)d_in[21];

  char* ws = (char*)d_ws;
  size_t off = 0;
  auto alloc = [&](size_t bytes) {
    void* p = ws + off; off += (bytes + 255) & ~size_t(255); return p;
  };
  float*  x    = (float*) alloc((size_t)M_SZ * D_SZ * 4);
  bf16_t* lnb_ = (bf16_t*)alloc((size_t)M_SZ * D_SZ * 2);
  bf16_t* qbuf = (bf16_t*)alloc((size_t)M_SZ * D_SZ * 2);
  bf16_t* kbuf = (bf16_t*)alloc((size_t)M_SZ * KVD_SZ * 2);
  bf16_t* vbuf = (bf16_t*)alloc((size_t)M_SZ * KVD_SZ * 2);
  bf16_t* kTb  = (bf16_t*)alloc((size_t)2 * G_SZ * HD_SZ * S_SZ * 2);
  bf16_t* attb = (bf16_t*)alloc((size_t)M_SZ * D_SZ * 2);
  bf16_t* h1b  = (bf16_t*)alloc((size_t)M_SZ * FF_SZ * 2);
  bf16_t* wteT = (bf16_t*)alloc((size_t)D_SZ * WTET_LD * 2);

  float* logits = (float*)d_out;
  float* lossp  = logits + (size_t)M_SZ * V_SZ;

  k_wteT<<<dim3((WTET_LD + 31) / 32, D_SZ / 32), dim3(32, 8), 0, stream>>>(wte, wteT);
  k_embed<<<M_SZ, 256, 0, stream>>>(idx, wte, wpe, x);

  for (int L = 0; L < NL_SZ; L++) {
    k_layernorm<<<M_SZ, 256, 0, stream>>>(x, ln1g + L * D_SZ, ln1b + L * D_SZ, lnb_);
    k_gemm<EPI_BF16, false><<<dim3(D_SZ / 128, M_SZ / 128), 256, 0, stream>>>(
        lnb_, D_SZ, wq + (size_t)L * D_SZ * D_SZ, D_SZ, bq + L * D_SZ,
        nullptr, 0, qbuf, D_SZ, D_SZ, D_SZ);
    k_gemm<EPI_BF16, false><<<dim3(KVD_SZ / 128, M_SZ / 128), 256, 0, stream>>>(
        lnb_, D_SZ, wk + (size_t)L * D_SZ * KVD_SZ, KVD_SZ, bk + L * KVD_SZ,
        nullptr, 0, kbuf, KVD_SZ, KVD_SZ, D_SZ);
    k_gemm<EPI_BF16, false><<<dim3(KVD_SZ / 128, M_SZ / 128), 256, 0, stream>>>(
        lnb_, D_SZ, wv + (size_t)L * D_SZ * KVD_SZ, KVD_SZ, bv + L * KVD_SZ,
        nullptr, 0, vbuf, KVD_SZ, KVD_SZ, D_SZ);
    k_transpose_k<<<M_SZ, 256, 0, stream>>>(kbuf, kTb);
    k_attn<<<2 * H_SZ * (S_SZ / 128), 256, 0, stream>>>(qbuf, kTb, vbuf, attb);
    k_gemm<EPI_RES, false><<<dim3(D_SZ / 128, M_SZ / 128), 256, 0, stream>>>(
        attb, D_SZ, wo + (size_t)L * D_SZ * D_SZ, D_SZ, bo + L * D_SZ,
        x, D_SZ, x, D_SZ, D_SZ, D_SZ);
    k_layernorm<<<M_SZ, 256, 0, stream>>>(x, ln2g + L * D_SZ, ln2b + L * D_SZ, lnb_);
    k_gemm<EPI_GELU, false><<<dim3(FF_SZ / 128, M_SZ / 128), 256, 0, stream>>>(
        lnb_, D_SZ, w1 + (size_t)L * D_SZ * FF_SZ, FF_SZ, b1 + L * FF_SZ,
        nullptr, 0, h1b, FF_SZ, FF_SZ, D_SZ);
    k_gemm<EPI_RES, false><<<dim3(D_SZ / 128, M_SZ / 128), 256, 0, stream>>>(
        h1b, FF_SZ, w2 + (size_t)L * FF_SZ * D_SZ, D_SZ, b2 + L * D_SZ,
        x, D_SZ, x, D_SZ, D_SZ, FF_SZ);
  }
  k_layernorm<<<M_SZ, 256, 0, stream>>>(x, lnfg, lnfb, lnb_);
  k_gemm<EPI_F32, true><<<dim3((V_SZ + 127) / 128, M_SZ / 128), 256, 0, stream>>>(
      lnb_, D_SZ, wteT, WTET_LD, nullptr, nullptr, 0, logits, V_SZ, V_SZ, D_SZ);
  hipMemsetAsync(lossp, 0, sizeof(float), stream);
  k_loss<<<M_SZ, 256, 0, stream>>>(logits, tgt, lossp);
}
